// MultiHeadAttn_37761352466662
// MI455X (gfx1250) — matrix operand order repaired and checked by reading, not compiled, not run on hardware
//
#include <hip/hip_runtime.h>
#include <hip/hip_bf16.h>

typedef __attribute__((ext_vector_type(16))) _Float16 v16h;
typedef __attribute__((ext_vector_type(8)))  _Float16 v8h;
typedef __attribute__((ext_vector_type(4)))  _Float16 v4h;
typedef __attribute__((ext_vector_type(8)))  float    v8f;

#define BB 2
#define NH 12
#define SS 2048
#define DM 768
#define DH 64
#define CW 128            // attention k-chunk width
#define NT (CW / 16)      // score tiles per chunk
#define LP 136            // padded LDS row stride (halves)

// ---- WMMA tile loaders (CDNA5 wave32 layouts, CDNA5 ISA §7.12.2) ----

// A-matrix 16x32 f16: lane L holds row L%16; half i -> K = k0 + (i<8?i:i+8) + (L>=16?8:0)
static __device__ inline v16h a_tile_f16(const _Float16* base, int ld, int k0, int lane) {
  const int l16 = lane & 15, hi = (lane >> 4) & 1;
  const _Float16* r = base + (size_t)l16 * ld + k0 + hi * 8;
  v8h lo = *(const v8h*)(r);
  v8h hh = *(const v8h*)(r + 16);
  v16h a;
#pragma unroll
  for (int i = 0; i < 8; ++i) { a[i] = lo[i]; a[8 + i] = hh[i]; }
  return a;
}

// B-matrix 32x16 f16: lane L holds column L%16; half i -> K = k0 + (i<8?i:i+8) + (L>=16?8:0)
static __device__ inline v16h b_tile_f16(const _Float16* base, int ld, int k0, int lane) {
  const int l16 = lane & 15, hi = (lane >> 4) & 1;
  const _Float16* r = base + (size_t)l16 * ld + k0 + hi * 8;
  v8h x0 = *(const v8h*)(r);
  v8h x1 = *(const v8h*)(r + 16);
  v16h b;
#pragma unroll
  for (int i = 0; i < 8; ++i) { b[i] = x0[i]; b[8 + i] = x1[i]; }
  return b;
}

static __device__ inline v8f wmma16(v16h a, v16h b, v8f c) {
  return __builtin_amdgcn_wmma_f32_16x16x32_f16(false, a, false, b, (short)0, c, false, false);
}

// ---- Kernel 0: one-shot f32 -> f16 conversion ------------------------------------
__global__ __launch_bounds__(256) void cvt_kernel(const float* __restrict__ s,
                                                  _Float16* __restrict__ d, int n4) {
  const int i = blockIdx.x * 256 + threadIdx.x;
  if (i < n4) {
    const float4 v = ((const float4*)s)[i];
    v4h h; h[0] = (_Float16)v.x; h[1] = (_Float16)v.y; h[2] = (_Float16)v.z; h[3] = (_Float16)v.w;
    *(v4h*)(d + 4 * (size_t)i) = h;
  }
}

// ---- Kernel 1: fused QKV projection (all-f16 operands), head split ---------------
// Q is pre-scaled by 1/sqrt(Dh)=0.125 (exact pow2) to delete the scale from attention.
// Q,K -> [B,H,S,Dh] f16 row-major; V -> [B,H,Dh,S] f16 (transposed)
__global__ __launch_bounds__(128) void qkv_proj_kernel(
    const _Float16* __restrict__ Xq, const _Float16* __restrict__ Xk, const _Float16* __restrict__ Xv,
    const _Float16* __restrict__ Wq, const _Float16* __restrict__ Wk, const _Float16* __restrict__ Wv,
    const float* __restrict__ Bq, const float* __restrict__ Bk, const float* __restrict__ Bv,
    _Float16* __restrict__ Qh, _Float16* __restrict__ Kh, _Float16* __restrict__ Vt) {
  const int lane = threadIdx.x & 31;
  const int gw = blockIdx.x * 4 + (threadIdx.x >> 5);   // global wave id
  const int m  = gw / (256 * NH);                       // 0=Q 1=K 2=V
  const int rr = gw % (256 * NH);
  const int i0 = (rr / NH) * 16;                        // row tile of 4096
  const int j0 = (rr % NH) * 64;                        // col tile of 768

  const _Float16* X  = (m == 0) ? Xq : (m == 1) ? Xk : Xv;
  const _Float16* W  = (m == 0) ? Wq : (m == 1) ? Wk : Wv;
  const float*    Bi = (m == 0) ? Bq : (m == 1) ? Bk : Bv;

  v8f acc[4] = {};
  for (int k0 = 0; k0 < DM; k0 += 32) {
    v16h a = a_tile_f16(X + (size_t)i0 * DM, DM, k0, lane);
#pragma unroll
    for (int t = 0; t < 4; ++t) {
      v16h b = b_tile_f16(W + (size_t)(j0 + t * 16) * DM, DM, k0, lane);
      acc[t] = wmma16(a, b, acc[t]);
    }
  }
  const int l16 = lane & 15, hi = (lane >> 4) & 1;
#pragma unroll
  for (int t = 0; t < 4; ++t) {
    const int col  = j0 + t * 16 + l16;
    const float bs = Bi[col];
    const int h = col / DH, d = col % DH;
#pragma unroll
    for (int j = 0; j < 8; ++j) {
      const int row = i0 + j + hi * 8;           // 0..4095
      const int bb = row / SS, s = row % SS;
      float v = acc[t][j] + bs;
      if (m == 0) v *= 0.125f;                   // fold softmax scale into Q
      if (m == 2) {
        Vt[(((size_t)bb * NH + h) * DH + d) * SS + s] = (_Float16)v;
      } else {
        _Float16* O = (m == 0) ? Qh : Kh;
        O[(((size_t)bb * NH + h) * SS + s) * DH + d] = (_Float16)v;
      }
    }
  }
}

// ---- Kernel 2: streaming causal attention (flash-style online softmax) ----------
// 1 wave = 16 q rows; block = 4 independent waves; 128-wide k chunks (32 WMMAs/chunk).
__global__ __launch_bounds__(128) void attn_kernel(
    const _Float16* __restrict__ Qh, const _Float16* __restrict__ Kh,
    const _Float16* __restrict__ Vt, _Float16* __restrict__ AO) {
  __shared__ __align__(16) _Float16 lds[4][16 * LP];    // wave-private P staging, padded stride
  const int lane = threadIdx.x & 31;
  const int w = threadIdx.x >> 5;
  const int l16 = lane & 15, hi = (lane >> 4) & 1;

  const int qb = blockIdx.x % (SS / 64);
  const int bh = blockIdx.x / (SS / 64);                // b*NH + h
  const int b  = bh / NH, h = bh % NH;
  const int q0 = qb * 64 + w * 16;

  const _Float16* Qbh = Qh + (size_t)bh * SS * DH;
  const _Float16* Kbh = Kh + (size_t)bh * SS * DH;
  const _Float16* Vbh = Vt + (size_t)bh * DH * SS;

  const v16h aq0 = a_tile_f16(Qbh + (size_t)q0 * DH, DH, 0, lane);
  const v16h aq1 = a_tile_f16(Qbh + (size_t)q0 * DH, DH, 32, lane);

  v8f o[4] = {};
  float rm[8], rs[8];
#pragma unroll
  for (int j = 0; j < 8; ++j) { rm[j] = -3.0e30f; rs[j] = 0.f; }

  for (int kb = 0; kb < q0 + 16; kb += CW) {
    // prefetch next K chunk (128 rows x 64 f16 = 16 KB) -> global_prefetch_b8
    if (kb + CW < q0 + 16)
      __builtin_prefetch(Kbh + (size_t)(kb + CW) * DH + lane * 256, 0, 1);

    // S = Q K^T over this 16x128 chunk (16 WMMAs); Q carries the 1/sqrt(Dh) scale
    v8f sc4[NT];
#pragma unroll
    for (int t = 0; t < NT; ++t) {
      v16h bk0 = b_tile_f16(Kbh + (size_t)(kb + t * 16) * DH, DH, 0, lane);
      v16h bk1 = b_tile_f16(Kbh + (size_t)(kb + t * 16) * DH, DH, 32, lane);
      v8f c = {};
      c = wmma16(aq0, bk0, c);
      c = wmma16(aq1, bk1, c);
      sc4[t] = c;
    }
    // causal mask only on diagonal chunks (interior chunks are mask-free)
    if (kb + CW - 1 > q0) {
#pragma unroll
      for (int t = 0; t < NT; ++t) {
        const int col = kb + t * 16 + l16;
#pragma unroll
        for (int j = 0; j < 8; ++j) {
          const int row = q0 + j + hi * 8;
          sc4[t][j] = (col <= row) ? sc4[t][j] : -3.0e30f;
        }
      }
    }
    // per-row chunk max
    float cmax[8];
#pragma unroll
    for (int j = 0; j < 8; ++j) cmax[j] = -3.0e30f;
#pragma unroll
    for (int t = 0; t < NT; ++t)
#pragma unroll
      for (int j = 0; j < 8; ++j) cmax[j] = fmaxf(cmax[j], sc4[t][j]);
#pragma unroll
    for (int j = 0; j < 8; ++j)
      for (int off = 1; off < 16; off <<= 1)
        cmax[j] = fmaxf(cmax[j], __shfl_xor(cmax[j], off, 32));

    // online-softmax rescale of running state
    float lsum[8];
#pragma unroll
    for (int j = 0; j < 8; ++j) {
      const float nm  = fmaxf(rm[j], cmax[j]);
      const float scl = __expf(rm[j] - nm);
      rm[j] = nm;
      rs[j] *= scl;
#pragma unroll
      for (int t = 0; t < 4; ++t) o[t][j] *= scl;
      lsum[j] = 0.f;
    }
    // P = exp(S - m), staged C-layout -> LDS (for A-layout reload)
#pragma unroll
    for (int t = 0; t < NT; ++t) {
#pragma unroll
      for (int j = 0; j < 8; ++j) {
        const float p = __expf(sc4[t][j] - rm[j]);
        lsum[j] += p;
        lds[w][(j + hi * 8) * LP + t * 16 + l16] = (_Float16)p;
      }
    }
#pragma unroll
    for (int j = 0; j < 8; ++j) {
      for (int off = 1; off < 16; off <<= 1)
        lsum[j] += __shfl_xor(lsum[j], off, 32);
      rs[j] += lsum[j];
    }
    asm volatile("s_wait_dscnt 0" ::: "memory");        // wave-private LDS RAW fence

    // O += P @ V (16 WMMAs); rows of V^T are exactly B-layout columns
#pragma unroll
    for (int kc = 0; kc < 4; ++kc) {
      v16h pa = a_tile_f16(&lds[w][0], LP, kc * 32, lane);
#pragma unroll
      for (int t = 0; t < 4; ++t) {
        v16h bv = b_tile_f16(Vbh + (size_t)(t * 16) * SS, SS, kb + kc * 32, lane);
        o[t] = wmma16(pa, bv, o[t]);
      }
    }
    asm volatile("s_wait_dscnt 0" ::: "memory");        // reads done before next-iter stores
  }

  // normalize and store attention output as f16 [B*S, DM]
#pragma unroll
  for (int t = 0; t < 4; ++t) {
#pragma unroll
    for (int j = 0; j < 8; ++j) {
      const int row = q0 + j + hi * 8;
      const float v = o[t][j] / rs[j];
      AO[((size_t)(b * SS + row)) * DM + h * DH + t * 16 + l16] = (_Float16)v;
    }
  }
}

// ---- Kernel 3: output projection (f16 x f16 -> f32 + bias) ----------------------
__global__ __launch_bounds__(128) void out_proj_kernel(
    const _Float16* __restrict__ AO, const _Float16* __restrict__ Wo,
    const float* __restrict__ Bo, float* __restrict__ Out) {
  const int lane = threadIdx.x & 31;
  const int gw = blockIdx.x * 4 + (threadIdx.x >> 5);
  const int i0 = (gw / NH) * 16, j0 = (gw % NH) * 64;
  v8f acc[4] = {};
  for (int k0 = 0; k0 < DM; k0 += 32) {
    v16h a = a_tile_f16(AO + (size_t)i0 * DM, DM, k0, lane);
#pragma unroll
    for (int t = 0; t < 4; ++t) {
      v16h b = b_tile_f16(Wo + (size_t)(j0 + t * 16) * DM, DM, k0, lane);
      acc[t] = wmma16(a, b, acc[t]);
    }
  }
  const int l16 = lane & 15, hi = (lane >> 4) & 1;
#pragma unroll
  for (int t = 0; t < 4; ++t) {
    const int col  = j0 + t * 16 + l16;
    const float bs = Bo[col];
#pragma unroll
    for (int j = 0; j < 8; ++j) {
      const int row = i0 + j + hi * 8;
      Out[(size_t)row * DM + col] = acc[t][j] + bs;
    }
  }
}

extern "C" void kernel_launch(void* const* d_in, const int* in_sizes, int n_in,
                              void* d_out, int out_size, void* d_ws, size_t ws_size,
                              hipStream_t stream) {
  (void)in_sizes; (void)n_in; (void)out_size; (void)ws_size;
  const float* Xq = (const float*)d_in[0];
  const float* Xk = (const float*)d_in[1];
  const float* Xv = (const float*)d_in[2];
  // d_in[3] = causal mask (bool) — applied analytically in-kernel
  const float* Wq = (const float*)d_in[4];
  const float* Bq = (const float*)d_in[5];
  const float* Wk = (const float*)d_in[6];
  const float* Bk = (const float*)d_in[7];
  const float* Wv = (const float*)d_in[8];
  const float* Bv = (const float*)d_in[9];
  const float* Wo = (const float*)d_in[10];
  const float* Bo = (const float*)d_in[11];

  const size_t xel = (size_t)BB * SS * DM;              // 3,145,728 (activations)
  const size_t wel = (size_t)DM * DM;                   // 589,824   (weights)
  _Float16* Qh  = (_Float16*)d_ws;                      // [B,H,S,Dh] (pre-scaled)
  _Float16* Kh  = Qh  + xel;                            // [B,H,S,Dh]
  _Float16* Vt  = Kh  + xel;                            // [B,H,Dh,S]
  _Float16* AO  = Vt  + xel;                            // [B*S, DM]
  _Float16* Xqh = AO  + xel;
  _Float16* Xkh = Xqh + xel;
  _Float16* Xvh = Xkh + xel;
  _Float16* Wqh = Xvh + xel;
  _Float16* Wkh = Wqh + wel;
  _Float16* Wvh = Wkh + wel;
  _Float16* Woh = Wvh + wel;

  // one-shot f32->f16 conversions (removes per-tile cvt from every GEMM k-loop)
  const int xn4 = (int)(xel / 4), wn4 = (int)(wel / 4);
  cvt_kernel<<<dim3((xn4 + 255) / 256), dim3(256), 0, stream>>>(Xq, Xqh, xn4);
  cvt_kernel<<<dim3((xn4 + 255) / 256), dim3(256), 0, stream>>>(Xk, Xkh, xn4);
  cvt_kernel<<<dim3((xn4 + 255) / 256), dim3(256), 0, stream>>>(Xv, Xvh, xn4);
  cvt_kernel<<<dim3((wn4 + 255) / 256), dim3(256), 0, stream>>>(Wq, Wqh, wn4);
  cvt_kernel<<<dim3((wn4 + 255) / 256), dim3(256), 0, stream>>>(Wk, Wkh, wn4);
  cvt_kernel<<<dim3((wn4 + 255) / 256), dim3(256), 0, stream>>>(Wv, Wvh, wn4);
  cvt_kernel<<<dim3((wn4 + 255) / 256), dim3(256), 0, stream>>>(Wo, Woh, wn4);

  qkv_proj_kernel<<<dim3(3 * 256 * NH / 4), dim3(128), 0, stream>>>(
      Xqh, Xkh, Xvh, Wqh, Wkh, Wvh, Bq, Bk, Bv, Qh, Kh, Vt);
  attn_kernel<<<dim3(BB * NH * (SS / 64)), dim3(128), 0, stream>>>(Qh, Kh, Vt, AO);
  out_proj_kernel<<<dim3(256 * NH / 4), dim3(128), 0, stream>>>(AO, Woh, Bo, (float*)d_out);
}
